// DotAttention_74406013436358
// MI455X (gfx1250) — matrix operand order repaired and checked by reading, not compiled, not run on hardware
//
#include <hip/hip_runtime.h>

typedef __bf16 v16bf __attribute__((ext_vector_type(16)));
typedef float  v8f   __attribute__((ext_vector_type(8)));

#define BATCH 4
#define TQL   4096
#define TVL   4096
#define DHEAD 64

#define WAVES 8
#define QROWS_PER_WAVE 16
#define QROWS_PER_BLOCK (WAVES * QROWS_PER_WAVE)   // 128
#define KV_TILE 32
#define NTILES (TVL / KV_TILE)                      // 128
#define TILE_ELEMS (KV_TILE * DHEAD)                // 2048 f32

__device__ __forceinline__ float rowReduceMax(float v) {
  v = fmaxf(v, __shfl_xor(v, 1, 32));
  v = fmaxf(v, __shfl_xor(v, 2, 32));
  v = fmaxf(v, __shfl_xor(v, 4, 32));
  v = fmaxf(v, __shfl_xor(v, 8, 32));
  return v;
}

// Async global->LDS copy of one 32x64 f32 tile (8 KB), 2 x b128 per thread.
// GVS mode: mem = SGPR64(base) + VGPR32(byte offset); LDS dest addr in VGPR.
__device__ __forceinline__ void asyncTileLoad(const float* __restrict__ base,
                                              float* ldsdst, int tid) {
  unsigned l0 = (unsigned)(uintptr_t)(ldsdst + tid * 4);
  unsigned g0 = (unsigned)(tid * 16);
  asm volatile("global_load_async_to_lds_b128 %0, %1, %2"
               :: "v"(l0), "v"(g0), "s"(base) : "memory");
  unsigned l1 = (unsigned)(uintptr_t)(ldsdst + (TILE_ELEMS / 2) + tid * 4);
  unsigned g1 = (unsigned)((TILE_ELEMS / 2) * 4 + tid * 16);
  asm volatile("global_load_async_to_lds_b128 %0, %1, %2"
               :: "v"(l1), "v"(g1), "s"(base) : "memory");
}

__global__ __launch_bounds__(WAVES * 32) void
DotAttention_74406013436358_kernel(const float* __restrict__ Q,
                                   const float* __restrict__ V,
                                   float* __restrict__ O) {
  // double-buffered raw f32 staging for async loads (16 KB)
  __shared__ __align__(16) float sStage[2][TILE_ELEMS];
  // V tile in two bf16 layouts for contiguous B-fragment ds_load_b128
  __shared__ __align__(32) __bf16 sV [KV_TILE * DHEAD];   // [kv][d]
  __shared__ __align__(32) __bf16 sVT[DHEAD * KV_TILE];   // [d][kv]
  // per-wave P scratch, row-major [m][k], for C-layout -> A-layout relayout
  __shared__ __align__(32) __bf16 sP [WAVES][QROWS_PER_WAVE * KV_TILE];

  const int tid  = threadIdx.x;
  const int wave = tid >> 5;
  const int lane = tid & 31;
  const int half = lane >> 4;   // 0: lanes 0-15, 1: lanes 16-31
  const int l16  = lane & 15;

  const int batch = blockIdx.x / (TQL / QROWS_PER_BLOCK);
  const int qtile = blockIdx.x % (TQL / QROWS_PER_BLOCK);
  const int qrow0 = qtile * QROWS_PER_BLOCK + wave * QROWS_PER_WAVE;

  const float* Qb = Q + (size_t)batch * TQL * DHEAD;
  const float* Vb = V + (size_t)batch * TVL * DHEAD;
  float*       Ob = O + (size_t)batch * TQL * DHEAD;

  // ---- Q A-fragments (16x32 bf16 x 2, covering D=64), resident in VGPRs ----
  // A-layout: lane holds row m = lane%16; element e -> K:
  //   K = 16*half + e (contiguous, matches B fragments)
  v16bf qa[2];
  {
    const float* qrow = Qb + (size_t)(qrow0 + l16) * DHEAD;
    for (int frag = 0; frag < 2; ++frag) {
      const int d0 = frag * 32;
      for (int e = 0; e < 16; ++e) {
        const int k = 16 * half + e;
        qa[frag][e] = (__bf16)qrow[d0 + k];
      }
    }
  }

  // all-ones B fragment: row sums of P via WMMA (P @ ones)
  v16bf ones;
  for (int e = 0; e < 16; ++e) ones[e] = (__bf16)1.0f;

  // ---- accumulators (C-layout: lane n=l16, VGPR j -> row m = j + 8*half) ----
  v8f acc[4];
  for (int i = 0; i < 4; ++i)
    for (int j = 0; j < 8; ++j) acc[i][j] = 0.0f;
  float rmax[8], rsum[8];
  for (int j = 0; j < 8; ++j) { rmax[j] = -__builtin_inff(); rsum[j] = 0.0f; }

  // prologue: kick off async fetch of tile 0
  asyncTileLoad(Vb, sStage[0], tid);

  for (int t = 0; t < NTILES; ++t) {
    const int cur = t & 1;
    const int nxt = cur ^ 1;

    // issue async fetch for next tile, then wait for current tile's 2 loads
    if (t + 1 < NTILES) {
      asyncTileLoad(Vb + (size_t)(t + 1) * TILE_ELEMS, sStage[nxt], tid);
      asm volatile("s_wait_asynccnt 0x2" ::: "memory");
    } else {
      asm volatile("s_wait_asynccnt 0x0" ::: "memory");
    }
    __syncthreads();  // tile t staged by all waves; prev compute done (sV reusable)

    // convert staged f32 tile -> bf16 in both layouts
    for (int i = 0; i < TILE_ELEMS / (WAVES * 32); ++i) {
      const int flat = i * (WAVES * 32) + tid;
      const int kv   = flat >> 6;
      const int d    = flat & 63;
      const __bf16 b = (__bf16)sStage[cur][flat];
      sV [kv * DHEAD + d]   = b;
      sVT[d * KV_TILE + kv] = b;
    }
    __syncthreads();  // sV/sVT ready for all waves

    // ---- S = Q @ Vtile^T : two 16x16 subtiles over n ----
    v8f S[2];
    for (int nsub = 0; nsub < 2; ++nsub) {
      v8f c;
      for (int j = 0; j < 8; ++j) c[j] = 0.0f;
      for (int frag = 0; frag < 2; ++frag) {
        const int d0 = frag * 32;
        // B[k][n] = V[nsub*16 + n][d0 + k]; lane n=l16, K = e + 16*half -> 16 contiguous bf16
        const __bf16* src = &sV[(nsub * 16 + l16) * DHEAD + d0 + 16 * half];
        v16bf b = *(const v16bf*)src;
        c = __builtin_amdgcn_wmma_f32_16x16x32_bf16(false, qa[frag], false, b,
                                                    (short)0, c, false, false);
      }
      S[nsub] = c;
    }

    // ---- online softmax: running max (shuffle butterfly within 16-lane half) ----
    float alpha[8];
    for (int j = 0; j < 8; ++j) {
      const float mloc = fmaxf(S[0][j], S[1][j]);
      const float mrow = rowReduceMax(mloc);
      const float mnew = fmaxf(rmax[j], mrow);
      alpha[j] = __expf(rmax[j] - mnew);
      S[0][j]  = __expf(S[0][j] - mnew);
      S[1][j]  = __expf(S[1][j] - mnew);
      rmax[j]  = mnew;
    }
    for (int i = 0; i < 4; ++i)
      for (int j = 0; j < 8; ++j) acc[i][j] *= alpha[j];

    // ---- P relayout C->A via per-wave LDS scratch (DS ops in-order per wave) ----
    __bf16* p = sP[wave];
    for (int nsub = 0; nsub < 2; ++nsub)
      for (int j = 0; j < 8; ++j) {
        const int m = j + 8 * half;
        const int k = nsub * 16 + l16;
        p[m * KV_TILE + k] = (__bf16)S[nsub][j];
      }
    v16bf pa;
    for (int e = 0; e < 16; ++e) {
      const int k = 16 * half + e;
      pa[e] = p[l16 * KV_TILE + k];
    }

    // ---- row sums via WMMA: srow = P @ ones (replaces shuffle-sum tree) ----
    {
      v8f z;
      for (int j = 0; j < 8; ++j) z[j] = 0.0f;
      v8f srow = __builtin_amdgcn_wmma_f32_16x16x32_bf16(false, pa, false, ones,
                                                         (short)0, z, false, false);
      for (int j = 0; j < 8; ++j) rsum[j] = rsum[j] * alpha[j] + srow[j];
    }

    // ---- acc += P @ Vtile : four 16-col d subtiles, K = 32 ----
    for (int dsub = 0; dsub < 4; ++dsub) {
      // B[k][n] = V[k][dsub*16 + n]; from sVT: lane n=l16, K = e + 16*half contiguous
      const __bf16* src = &sVT[(dsub * 16 + l16) * KV_TILE + 16 * half];
      v16bf b = *(const v16bf*)src;
      acc[dsub] = __builtin_amdgcn_wmma_f32_16x16x32_bf16(false, pa, false, b,
                                                          (short)0, acc[dsub],
                                                          false, false);
    }
  }

  // ---- normalize and store (C-layout scatter, dword stores) ----
  for (int dsub = 0; dsub < 4; ++dsub) {
    for (int j = 0; j < 8; ++j) {
      const int m = j + 8 * half;
      const int d = dsub * 16 + l16;
      Ob[(size_t)(qrow0 + m) * DHEAD + d] = acc[dsub][j] / rsum[j];
    }
  }
}

extern "C" void kernel_launch(void* const* d_in, const int* in_sizes, int n_in,
                              void* d_out, int out_size, void* d_ws, size_t ws_size,
                              hipStream_t stream) {
  const float* Q = (const float*)d_in[0];
  const float* V = (const float*)d_in[1];
  float* out = (float*)d_out;
  (void)in_sizes; (void)n_in; (void)out_size; (void)d_ws; (void)ws_size;

  dim3 grid(BATCH * (TQL / QROWS_PER_BLOCK));  // 4 * 32 = 128 blocks
  dim3 block(WAVES * 32);                      // 256 threads = 8 waves (wave32)
  hipLaunchKernelGGL(DotAttention_74406013436358_kernel, grid, block, 0, stream,
                     Q, V, out);
}
